// QLSTM_65481071402381
// MI455X (gfx1250) — matrix operand order repaired and checked by reading, not compiled, not run on hardware
//
#include <hip/hip_runtime.h>
#include <hip/hip_bf16.h>

#define TT 512
#define BN 64
#define DN 256
#define HN 512
#define KN 768          // DN + HN
#define NWG 32          // each workgroup owns 16 hidden columns (x4 gates)
#define THREADS 256     // 8 waves of 32
#define KPAD 16
#define SROW (KN + KPAD)   // 784 halves: row stride 1568B -> banks shift 8/row
#define WROW (KN + KPAD)   // same padding for the weight tile

typedef __attribute__((ext_vector_type(16))) _Float16 v16h;
typedef __attribute__((ext_vector_type(8)))  _Float16 v8h;
typedef __attribute__((ext_vector_type(8)))  float    v8f;

__device__ __forceinline__ v16h cat8(v8h lo, v8h hi) {
  return __builtin_shufflevector(lo, hi, 0,1,2,3,4,5,6,7,8,9,10,11,12,13,14,15);
}

__device__ __forceinline__ float sigmoidf_(float x) {
  return 1.0f / (1.0f + __expf(-x));
}
__device__ __forceinline__ float tanhf_(float x) {   // branchless: 2*sigma(2x)-1
  return __builtin_fmaf(2.0f, sigmoidf_(2.0f * x), -1.0f);
}

__global__ __launch_bounds__(THREADS, 1)
void qlstm_persistent(const float* __restrict__ X,      // [TT][BN][DN] fp32
                      const float* __restrict__ Wf, const float* __restrict__ Wi,
                      const float* __restrict__ Wg, const float* __restrict__ Wo,
                      const float* __restrict__ bfv, const float* __restrict__ biv,
                      const float* __restrict__ bgv, const float* __restrict__ bov,
                      _Float16* __restrict__ h16,       // [2][BN][HN] f16 (ping-pong)
                      unsigned* __restrict__ bar,       // [0]=count [1]=generation
                      float* __restrict__ out,          // [TT][BN][HN] fp32
                      float* __restrict__ hx,           // [BN][HN]
                      float* __restrict__ cx)           // [BN][HN]
{
  __shared__ __align__(16) _Float16 sComb[BN][SROW];     // ~100 KB activations
  __shared__ __align__(16) _Float16 sW[4][16][WROW];     // ~100 KB resident weights
  __shared__ float sZ[4][BN][20];                        // ~20 KB gate pre-acts

  const int tid   = threadIdx.x;
  const int wg    = blockIdx.x;
  const int n0    = wg * 16;
  const int wave  = tid >> 5;
  const int lane  = tid & 31;
  const int laneN = lane & 15;
  const int laneG = lane >> 4;
  const int gate  = wave & 3;
  const int mt0   = (wave >> 2) * 2;   // this wave's two M-tiles

  // ---- one-time: stage this WG's weight slice into LDS (f32 -> f16) ----
  for (int c = tid; c < 4 * 16 * (KN / 8); c += THREADS) {
    int g    = c / (16 * (KN / 8));
    int rem  = c % (16 * (KN / 8));
    int r    = rem / (KN / 8);
    int col  = (rem % (KN / 8)) * 8;
    const float* Wsrc = (g == 0) ? Wf : (g == 1) ? Wi : (g == 2) ? Wg : Wo;
    const float* src = Wsrc + (size_t)(n0 + r) * KN + col;
    v8h v;
    #pragma unroll
    for (int j = 0; j < 8; ++j) v[j] = (_Float16)src[j];
    *(v8h*)&sW[g][r][col] = v;
  }

  // ---- per-thread persistent state: cell values + biases in registers ----
  float creg[4], bfr[4], bir[4], bgr[4], bor[4];
  #pragma unroll
  for (int i = 0; i < 4; ++i) {
    int n = n0 + ((tid + THREADS * i) & 15);
    creg[i] = 0.0f;
    bfr[i] = bfv[n]; bir[i] = biv[n]; bgr[i] = bgv[n]; bor[i] = bov[n];
  }

  const _Float16* browLds = &sW[gate][laneN][8 * laneG];
  const int ra0 = mt0 * 16 + laneN;
  const int ra1 = ra0 + 16;

  __syncthreads();

  for (int t = 0; t < TT; ++t) {
    // ---- stage comb = [x_t | h_{t-1}] into LDS as f16 ----
    const float*    xt    = X   + (size_t)t * BN * DN;
    const _Float16* hprev = h16 + (size_t)(t & 1) * BN * HN;
    for (int c = tid; c < BN * (KN / 8); c += THREADS) {
      int row = c / (KN / 8);
      int col = (c % (KN / 8)) * 8;
      v8h v;
      if (col < DN) {
        const float* src = xt + row * DN + col;
        #pragma unroll
        for (int j = 0; j < 8; ++j) v[j] = (_Float16)src[j];
      } else {
        v = *(const v8h*)(hprev + row * HN + (col - DN));
      }
      *(v8h*)&sComb[row][col] = v;
    }
    if (t + 1 < TT)   // pull next timestep's x toward the WGP while we compute
      __builtin_prefetch(X + (size_t)(t + 1) * BN * DN + tid * 64, 0, 1);
    __syncthreads();

    // ---- GEMM: two 16x16 tiles of this wave's gate, K = 768, all from LDS ----
    v8f acc0 = {}; v8f acc1 = {};
    #pragma unroll 4
    for (int kk = 0; kk < KN / 32; ++kk) {
      const int k0 = kk * 32;
      v16h bm = cat8(*(const v8h*)(browLds + k0),
                     *(const v8h*)(browLds + k0 + 16));
      v16h a0 = cat8(*(const v8h*)&sComb[ra0][k0 + 8 * laneG],
                     *(const v8h*)&sComb[ra0][k0 + 16 + 8 * laneG]);
      v16h a1 = cat8(*(const v8h*)&sComb[ra1][k0 + 8 * laneG],
                     *(const v8h*)&sComb[ra1][k0 + 16 + 8 * laneG]);
      acc0 = __builtin_amdgcn_wmma_f32_16x16x32_f16(false, a0, false, bm, (short)0, acc0, false, false);
      acc1 = __builtin_amdgcn_wmma_f32_16x16x32_f16(false, a1, false, bm, (short)0, acc1, false, false);
    }
    // C/D layout: VGPR r -> row r + 8*(lane/16); lane%16 -> column
    #pragma unroll
    for (int r = 0; r < 8; ++r) {
      sZ[gate][mt0 * 16 + r + 8 * laneG][laneN]       = acc0[r];
      sZ[gate][(mt0 + 1) * 16 + r + 8 * laneG][laneN] = acc1[r];
    }
    __syncthreads();

    // ---- fused gate math; c and biases live in registers ----
    _Float16* hnext = h16 + (size_t)((t + 1) & 1) * BN * HN;
    #pragma unroll
    for (int i = 0; i < 4; ++i) {
      int e   = tid + THREADS * i;   // 0..1023
      int b   = e >> 4;
      int col = e & 15;
      float fg = sigmoidf_(sZ[0][b][col] + bfr[i]);
      float ig = sigmoidf_(sZ[1][b][col] + bir[i]);
      float gg = tanhf_   (sZ[2][b][col] + bgr[i]);
      float og = sigmoidf_(sZ[3][b][col] + bor[i]);
      float cv = __builtin_fmaf(fg, creg[i], ig * gg);
      creg[i] = cv;
      float hv = og * tanhf_(cv);
      size_t ci = (size_t)b * HN + (n0 + col);
      out[(size_t)t * BN * HN + ci] = hv;
      hnext[ci] = (_Float16)hv;
      if (t == TT - 1) { hx[ci] = hv; cx[ci] = cv; }
    }

    // ---- device-wide generation barrier before the next step ----
    __syncthreads();
    __threadfence();
    if (tid == 0) {
      volatile unsigned* vgen = bar + 1;
      unsigned oldg = *vgen;
      if (atomicAdd(bar, 1u) == NWG - 1) {
        atomicExch(bar, 0u);
        __threadfence();
        atomicAdd(bar + 1, 1u);
      } else {
        while (*vgen == oldg) __builtin_amdgcn_s_sleep(2);
      }
    }
    __syncthreads();
  }
}

extern "C" void kernel_launch(void* const* d_in, const int* in_sizes, int n_in,
                              void* d_out, int out_size, void* d_ws, size_t ws_size,
                              hipStream_t stream) {
  const float* X   = (const float*)d_in[0];
  const float* Wf  = (const float*)d_in[1];
  const float* bf_ = (const float*)d_in[2];
  const float* Wi  = (const float*)d_in[3];
  const float* bi_ = (const float*)d_in[4];
  const float* Wg  = (const float*)d_in[5];
  const float* bg_ = (const float*)d_in[6];
  const float* Wo  = (const float*)d_in[7];
  const float* bo_ = (const float*)d_in[8];

  char* ws = (char*)d_ws;
  _Float16* h16 = (_Float16*)(ws);            // 2*BN*HN f16 = 131072 B
  unsigned* bar = (unsigned*)(ws + 131072);   // 8 B barrier state

  // deterministic init: zero h(0) ping-pong buffers + barrier state
  hipMemsetAsync(d_ws, 0, 131072 + 256, stream);

  float* out = (float*)d_out;
  float* hx  = out + (size_t)TT * BN * HN;
  float* cx  = hx + (size_t)BN * HN;
  qlstm_persistent<<<NWG, THREADS, 0, stream>>>(X, Wf, Wi, Wg, Wo,
                                                bf_, bi_, bg_, bo_,
                                                h16, bar, out, hx, cx);
}
